// STMambaBlock_90692529422495
// MI455X (gfx1250) — matrix operand order repaired and checked by reading, not compiled, not run on hardware
//
#include <hip/hip_runtime.h>
#include <hip/hip_bf16.h>

// ---------------------------------------------------------------------------
// STMambaBlock for MI455X (gfx1250, wave32, WMMA).
// All dense projections run on v_wmma_f32_16x16x32_f16 (f16 in, f32 acc).
// GEMM uses 4x2 register blocking per wave (64x32 output) for ~21 flops/byte.
// Every GEMM in this model has M = 24576 tokens; N is padded to mult of 32.
// ---------------------------------------------------------------------------

typedef __attribute__((ext_vector_type(16))) _Float16 v16h;
typedef __attribute__((ext_vector_type(8)))  float    v8f;

static constexpr int CDIM = 128;   // model dim
static constexpr int DIN  = 256;   // inner dim
static constexpr int DST  = 16;    // state dim
static constexpr int DTR  = 8;     // dt rank
static constexpr int BB = 2, TT = 12, HH = 32, WWD = 32;
static constexpr int HWC = HH * WWD;     // 1024
static constexpr int BT  = BB * TT;      // 24
static constexpr int BHW = BB * HWC;     // 2048
static constexpr int TOK = 24576;        // tokens for every GEMM (2048*12 == 24*1024)
static constexpr int XPD = 64;           // xproj rows padded (40 -> 64, zeros)
static constexpr int GX  = TOK / 256;    // GEMM grid.x (4 waves x 64 rows per block)

static __device__ __forceinline__ float sigmoidf_(float x) { return 1.f / (1.f + __expf(-x)); }
static __device__ __forceinline__ float siluf_(float x)    { return x * sigmoidf_(x); }

// ---------------------------------------------------------------------------
// WMMA GEMM:  C[M,N] = A[M,K] * W[N,K]^T   (A,W f16 row-major; C f32)
// block = 128 threads = 4 waves; each wave computes a 64x32 output block
// (4 M-tiles x 2 N-tiles), reusing each A fragment for 2 WMMAs and each B
// fragment for 4 WMMAs.  grid = (M/256, N/32).  K % 32 == 0.
// ---------------------------------------------------------------------------
__global__ void gemm_wmma_f16_kernel(const _Float16* __restrict__ A,
                                     const _Float16* __restrict__ W,
                                     float* __restrict__ C,
                                     int M, int N, int K)
{
    const int wave = threadIdx.x >> 5;
    const int lane = threadIdx.x & 31;
    const int mlo  = lane & 15;
    const int hi   = lane >> 4;            // K-half select per ISA A/B layouts
    const int rowBase = (blockIdx.x * 4 + wave) * 64;
    const int colBase = blockIdx.y * 32;

    const _Float16* arow[4];
    #pragma unroll
    for (int i = 0; i < 4; ++i)
        arow[i] = A + (size_t)(rowBase + i * 16 + mlo) * K + hi * 8;
    const _Float16* wrow[2];
    #pragma unroll
    for (int j = 0; j < 2; ++j)
        wrow[j] = W + (size_t)(colBase + j * 16 + mlo) * K + hi * 8;

    v8f acc[4][2] = {};
    for (int k0 = 0; k0 < K; k0 += 32) {
        // B fragments: column n = lane&15; same 8-wide K interleave as A.
        v16h b0, b1;
        {
            const _Float16* bp0 = wrow[0] + k0;
            const _Float16* bp1 = wrow[1] + k0;
            #pragma unroll
            for (int h = 0; h < 8; ++h) { b0[h] = bp0[h]; b0[8 + h] = bp0[16 + h]; b1[h] = bp1[h]; b1[8 + h] = bp1[16 + h]; }
        }
        #pragma unroll
        for (int i = 0; i < 4; ++i) {
            // A fragment: 16-bit A 16x32 layout (8-wide K interleave).
            v16h a;
            const _Float16* ap = arow[i] + k0;
            #pragma unroll
            for (int t = 0; t < 8; ++t) { a[t] = ap[t]; a[8 + t] = ap[16 + t]; }
            __builtin_prefetch((const void*)(ap + 32), 0, 1);  // next K block
            acc[i][0] = __builtin_amdgcn_wmma_f32_16x16x32_f16(
                            false, a, false, b0, (short)0, acc[i][0], false, false);
            acc[i][1] = __builtin_amdgcn_wmma_f32_16x16x32_f16(
                            false, a, false, b1, (short)0, acc[i][1], false, false);
        }
    }
    // D layout: lanes 0-15 -> rows 0..7, lanes 16-31 -> rows 8..15.
    #pragma unroll
    for (int i = 0; i < 4; ++i) {
        #pragma unroll
        for (int j = 0; j < 2; ++j) {
            float* crow = C + (size_t)(rowBase + i * 16 + hi * 8) * N
                            + (colBase + j * 16 + mlo);
            #pragma unroll
            for (int r = 0; r < 8; ++r) crow[(size_t)r * N] = acc[i][j][r];
        }
    }
}

// ---------------------------------------------------------------------------
// LayerNorm over channels fused with the (B,T,C,H,W)->(B*H*W, T, C) transpose.
// One wave per token; 4 channels per lane; wave32 shuffle reduction.
// ---------------------------------------------------------------------------
__global__ void ln_temporal_kernel(const float* __restrict__ x,
                                   const float* __restrict__ g,
                                   const float* __restrict__ bt,
                                   _Float16* __restrict__ out)
{
    const int token = blockIdx.x * 8 + (threadIdx.x >> 5);   // grid exact
    const int lane  = threadIdx.x & 31;
    const int n  = token / TT, t = token - n * TT;
    const int b  = n / HWC,  hw = n - b * HWC;
    const float* base = x + ((size_t)(b * TT + t) * CDIM) * HWC + hw;
    float v[4], s = 0.f;
    #pragma unroll
    for (int j = 0; j < 4; ++j) { v[j] = base[(size_t)(lane + 32 * j) * HWC]; s += v[j]; }
    #pragma unroll
    for (int m = 16; m >= 1; m >>= 1) s += __shfl_xor(s, m, 32);
    const float mean = s * (1.f / 128.f);
    float var = 0.f;
    #pragma unroll
    for (int j = 0; j < 4; ++j) { float d = v[j] - mean; var += d * d; }
    #pragma unroll
    for (int m = 16; m >= 1; m >>= 1) var += __shfl_xor(var, m, 32);
    const float inv = rsqrtf(var * (1.f / 128.f) + 1e-5f);
    _Float16* orow = out + (size_t)token * CDIM;
    #pragma unroll
    for (int j = 0; j < 4; ++j) {
        int c = lane + 32 * j;
        orow[c] = (_Float16)((v[j] - mean) * inv * g[c] + bt[c]);
    }
}

// Spatial LN: reads xw (B,T,C,H,W); writes fwd tokens and L-flipped tokens.
__global__ void ln_spatial_kernel(const float* __restrict__ xw,
                                  const float* __restrict__ g,
                                  const float* __restrict__ bt,
                                  _Float16* __restrict__ outF,
                                  _Float16* __restrict__ outB)
{
    const int token = blockIdx.x * 8 + (threadIdx.x >> 5);
    const int lane  = threadIdx.x & 31;
    const int btk = token / HWC, hw = token - btk * HWC;
    const float* base = xw + ((size_t)btk * CDIM) * HWC + hw;
    float v[4], s = 0.f;
    #pragma unroll
    for (int j = 0; j < 4; ++j) { v[j] = base[(size_t)(lane + 32 * j) * HWC]; s += v[j]; }
    #pragma unroll
    for (int m = 16; m >= 1; m >>= 1) s += __shfl_xor(s, m, 32);
    const float mean = s * (1.f / 128.f);
    float var = 0.f;
    #pragma unroll
    for (int j = 0; j < 4; ++j) { float d = v[j] - mean; var += d * d; }
    #pragma unroll
    for (int m = 16; m >= 1; m >>= 1) var += __shfl_xor(var, m, 32);
    const float inv = rsqrtf(var * (1.f / 128.f) + 1e-5f);
    _Float16* oF = outF + (size_t)token * CDIM;
    _Float16* oB = outB + ((size_t)btk * HWC + (HWC - 1 - hw)) * CDIM;
    #pragma unroll
    for (int j = 0; j < 4; ++j) {
        int c = lane + 32 * j;
        _Float16 h = (_Float16)((v[j] - mean) * inv * g[c] + bt[c]);
        oF[c] = h; oB[c] = h;
    }
}

// Depthwise causal conv (len 4) + bias + silu on first 256 cols of xz.
__global__ void dwconv_silu_kernel(const float* __restrict__ xz,
                                   const float* __restrict__ cw,
                                   const float* __restrict__ cb,
                                   float* __restrict__ xc,
                                   _Float16* __restrict__ xc16,
                                   int Nn, int L)
{
    const int idx = blockIdx.x * 256 + threadIdx.x;       // over Nn*L*256 (exact)
    const int c   = idx & 255;
    const int row = idx >> 8;
    const int l   = row % L;
    const int n   = row / L;
    float acc = cb[c];
    #pragma unroll
    for (int j = 0; j < 4; ++j) {
        int ll = l - 3 + j;
        if (ll >= 0) acc += cw[c * 4 + j] * xz[(size_t)(n * L + ll) * 512 + c];
    }
    float s = siluf_(acc);
    xc[idx]   = s;
    xc16[idx] = (_Float16)s;
}

// dt = softplus(dbl[:, :8] @ dt_w^T + dt_b)    (dbl stride = XPD)
__global__ void dtproj_kernel(const float* __restrict__ dbl,
                              const float* __restrict__ dt_w,
                              const float* __restrict__ dt_b,
                              float* __restrict__ dt)
{
    const int idx = blockIdx.x * 256 + threadIdx.x;       // over TOK*256
    const int c   = idx & 255;
    const int row = idx >> 8;
    float s = dt_b[c];
    const float* dr = dbl + (size_t)row * XPD;
    #pragma unroll
    for (int r = 0; r < DTR; ++r) s += dr[r] * dt_w[c * DTR + r];
    dt[idx] = (s > 20.f) ? s : log1pf(__expf(s));
}

// Selective scan: 16 lanes = one (n,d) pair, one state element per lane.
__global__ void scan_kernel(const float* __restrict__ dt,
                            const float* __restrict__ xc,
                            const float* __restrict__ dbl,
                            const float* __restrict__ Alog,
                            float* __restrict__ y,
                            int Nn, int L)
{
    const int tid = blockIdx.x * 256 + threadIdx.x;       // over Nn*256*16 (exact)
    const int s = tid & 15;
    const int d = (tid >> 4) & 255;
    const int n = tid >> 12;
    const float a = -__expf(Alog[d * DST + s]);
    float h = 0.f;
    for (int l = 0; l < L; ++l) {
        const size_t row = (size_t)(n * L + l);
        const float dtv = dt[row * 256 + d];
        const float xv  = xc[row * 256 + d];
        const float bv  = dbl[row * XPD + DTR + s];
        const float cv  = dbl[row * XPD + DTR + DST + s];
        h = __expf(dtv * a) * h + (dtv * xv) * bv;
        float p = h * cv;
        p += __shfl_xor(p, 1, 32);
        p += __shfl_xor(p, 2, 32);
        p += __shfl_xor(p, 4, 32);
        p += __shfl_xor(p, 8, 32);
        if (s == 0) y[row * 256 + d] = p;
    }
}

// y = (scan + xc*D) * silu(z)  -> f16 for the out-projection GEMM.
__global__ void gate_kernel(const float* __restrict__ y,
                            const float* __restrict__ xc,
                            const float* __restrict__ xz,
                            const float* __restrict__ Dp,
                            _Float16* __restrict__ gate16)
{
    const int idx = blockIdx.x * 256 + threadIdx.x;       // over TOK*256
    const int d   = idx & 255;
    const int row = idx >> 8;
    const float g = y[idx] + xc[idx] * Dp[d];
    const float z = xz[(size_t)row * 512 + 256 + d];
    gate16[idx] = (_Float16)(g * siluf_(z));
}

// x += temporal mamba output (token layout -> (B,T,C,H,W))
__global__ void residual_t_kernel(const float* __restrict__ x,
                                  const float* __restrict__ mout,
                                  float* __restrict__ xres)
{
    const int idx = blockIdx.x * 256 + threadIdx.x;       // over B*T*C*HW
    const int hw = idx & (HWC - 1);
    const int c  = (idx >> 10) & (CDIM - 1);
    const int t  = (idx >> 17) % TT;
    const int b  = idx / (HWC * CDIM * TT);
    const int row = (b * HWC + hw) * TT + t;
    xres[idx] = x[idx] + mout[(size_t)row * CDIM + c];
}

// 3x3 conv 128->32, pad 1, + bias + silu.
__global__ void conv1_kernel(const float* __restrict__ xres,
                             const float* __restrict__ w,
                             const float* __restrict__ b,
                             float* __restrict__ gbuf)
{
    const int idx = blockIdx.x * 256 + threadIdx.x;       // over BT*32*HW
    const int hw = idx & (HWC - 1);
    const int oc = (idx >> 10) & 31;
    const int bt = idx >> 15;
    const int h0 = hw >> 5, w0 = hw & 31;
    float acc = b[oc];
    const float* ib = xres + ((size_t)bt * CDIM) * HWC;
    const float* wb = w + ((size_t)oc * CDIM) * 9;
    for (int c = 0; c < CDIM; ++c) {
        const float* ip = ib + (size_t)c * HWC;
        const float* wp = wb + c * 9;
        #pragma unroll
        for (int kh = 0; kh < 3; ++kh) {
            int hy = h0 + kh - 1;
            if (hy < 0 || hy >= HH) continue;
            #pragma unroll
            for (int kw = 0; kw < 3; ++kw) {
                int wx = w0 + kw - 1;
                if (wx < 0 || wx >= WWD) continue;
                acc += wp[kh * 3 + kw] * ip[hy * WWD + wx];
            }
        }
    }
    gbuf[idx] = siluf_(acc);
}

// 1x1 conv 32->1 + sigmoid; apply importance scale to xres -> xw.
__global__ void gate2_apply_kernel(const float* __restrict__ gbuf,
                                   const float* __restrict__ g2w,
                                   const float* __restrict__ g2b,
                                   const float* __restrict__ xres,
                                   float* __restrict__ xw)
{
    const int idx = blockIdx.x * 256 + threadIdx.x;       // over BT*HW
    const int hw = idx & (HWC - 1);
    const int bt = idx >> 10;
    float s = g2b[0];
    #pragma unroll
    for (int oc = 0; oc < 32; ++oc)
        s += g2w[oc] * gbuf[((size_t)bt * 32 + oc) * HWC + hw];
    const float scale = 0.5f + 0.5f * sigmoidf_(s);
    for (int c = 0; c < CDIM; ++c) {
        size_t off = ((size_t)bt * CDIM + c) * HWC + hw;
        xw[off] = xres[off] * scale;
    }
}

// concat [of | unflip(ob)] -> f16 rows for final projection.
__global__ void concat_kernel(const float* __restrict__ of,
                              const float* __restrict__ ob,
                              _Float16* __restrict__ cat16)
{
    const int idx = blockIdx.x * 256 + threadIdx.x;       // over TOK*256
    const int col = idx & 255;
    const int row = idx >> 8;
    float v;
    if (col < CDIM) {
        v = of[(size_t)row * CDIM + col];
    } else {
        const int bt = row / HWC, hw = row - bt * HWC;
        const int fr = bt * HWC + (HWC - 1 - hw);
        v = ob[(size_t)fr * CDIM + (col - CDIM)];
    }
    cat16[idx] = (_Float16)v;
}

// out = xres + (proj + bias)   (token layout back to (B,T,C,H,W))
__global__ void final_kernel(const float* __restrict__ xres,
                             const float* __restrict__ proj,
                             const float* __restrict__ pb,
                             float* __restrict__ out)
{
    const int idx = blockIdx.x * 256 + threadIdx.x;       // over B*T*C*HW
    const int hw = idx & (HWC - 1);
    const int c  = (idx >> 10) & (CDIM - 1);
    const int bt = idx >> 17;
    const int row = bt * HWC + hw;
    out[idx] = xres[idx] + proj[(size_t)row * CDIM + c] + pb[c];
}

// weight conversion helpers
__global__ void cvt_f16_kernel(const float* __restrict__ s, _Float16* __restrict__ d, int n)
{
    const int i = blockIdx.x * 256 + threadIdx.x;
    if (i < n) d[i] = (_Float16)s[i];
}
__global__ void pad_xproj_kernel(const float* __restrict__ s, _Float16* __restrict__ d)
{
    const int i = blockIdx.x * 256 + threadIdx.x;         // over XPD*256
    const int r = i >> 8;
    d[i] = (r < 40) ? (_Float16)s[i] : (_Float16)0.f;
}

// ---------------------------------------------------------------------------
// Host orchestration
// ---------------------------------------------------------------------------
struct MambaW {
    const float *in_w, *conv_w, *conv_b, *xproj_w, *dt_w, *dt_b, *Alog, *D, *out_w;
};

static void run_mamba(const _Float16* tok, const _Float16* in_w16,
                      const _Float16* xproj16, const _Float16* out_w16,
                      const MambaW& w,
                      float* xz, float* xc, _Float16* xc16, float* dbl,
                      float* dt, float* ybuf, float* out,
                      int Nn, int L, hipStream_t stream)
{
    gemm_wmma_f16_kernel<<<dim3(GX, 512 / 32), 128, 0, stream>>>(
        tok, in_w16, xz, TOK, 512, CDIM);
    dwconv_silu_kernel<<<TOK * DIN / 256, 256, 0, stream>>>(
        xz, w.conv_w, w.conv_b, xc, xc16, Nn, L);
    gemm_wmma_f16_kernel<<<dim3(GX, XPD / 32), 128, 0, stream>>>(
        xc16, xproj16, dbl, TOK, XPD, DIN);
    dtproj_kernel<<<TOK * DIN / 256, 256, 0, stream>>>(dbl, w.dt_w, w.dt_b, dt);
    scan_kernel<<<(Nn * DIN * DST) / 256, 256, 0, stream>>>(
        dt, xc, dbl, w.Alog, ybuf, Nn, L);
    gate_kernel<<<TOK * DIN / 256, 256, 0, stream>>>(ybuf, xc, xz, w.D, xc16);
    gemm_wmma_f16_kernel<<<dim3(GX, CDIM / 32), 128, 0, stream>>>(
        xc16, out_w16, out, TOK, CDIM, DIN);
}

extern "C" void kernel_launch(void* const* d_in, const int* in_sizes, int n_in,
                              void* d_out, int out_size, void* d_ws, size_t ws_size,
                              hipStream_t stream)
{
    (void)in_sizes; (void)n_in; (void)out_size; (void)ws_size;
    const float* x = (const float*)d_in[0];
    MambaW mw[3];
    for (int m = 0; m < 3; ++m) {
        const int o = 1 + m * 9;
        mw[m].in_w    = (const float*)d_in[o + 0];
        mw[m].conv_w  = (const float*)d_in[o + 1];
        mw[m].conv_b  = (const float*)d_in[o + 2];
        mw[m].xproj_w = (const float*)d_in[o + 3];
        mw[m].dt_w    = (const float*)d_in[o + 4];
        mw[m].dt_b    = (const float*)d_in[o + 5];
        mw[m].Alog    = (const float*)d_in[o + 6];
        mw[m].D       = (const float*)d_in[o + 7];
        mw[m].out_w   = (const float*)d_in[o + 8];
    }
    const float* ln_t_g = (const float*)d_in[28];
    const float* ln_t_b = (const float*)d_in[29];
    const float* ln_s_g = (const float*)d_in[30];
    const float* ln_s_b = (const float*)d_in[31];
    const float* proj_w = (const float*)d_in[32];
    const float* proj_b = (const float*)d_in[33];
    const float* g1_w   = (const float*)d_in[34];
    const float* g1_b   = (const float*)d_in[35];
    const float* g2_w   = (const float*)d_in[36];
    const float* g2_b   = (const float*)d_in[37];
    float* out = (float*)d_out;

    // workspace layout (~250 MB)
    char* ws = (char*)d_ws;
    size_t off = 0;
    auto alloc = [&](size_t bytes) -> void* {
        void* p = ws + off; off += (bytes + 255) & ~(size_t)255; return p;
    };
    _Float16* in_w16[3];  _Float16* xproj16[3]; _Float16* out_w16[3];
    for (int m = 0; m < 3; ++m) {
        in_w16[m]  = (_Float16*)alloc((size_t)512 * CDIM * 2);
        xproj16[m] = (_Float16*)alloc((size_t)XPD * DIN * 2);
        out_w16[m] = (_Float16*)alloc((size_t)CDIM * DIN * 2);
    }
    _Float16* proj_w16 = (_Float16*)alloc((size_t)CDIM * DIN * 2);
    _Float16* tokA   = (_Float16*)alloc((size_t)TOK * CDIM * 2);
    _Float16* tokB   = (_Float16*)alloc((size_t)TOK * CDIM * 2);
    float*    xz     = (float*)alloc((size_t)TOK * 512 * 4);
    float*    xc     = (float*)alloc((size_t)TOK * DIN * 4);
    _Float16* xc16   = (_Float16*)alloc((size_t)TOK * DIN * 2);
    float*    dbl    = (float*)alloc((size_t)TOK * XPD * 4);
    float*    dtb    = (float*)alloc((size_t)TOK * DIN * 4);
    float*    ybuf   = (float*)alloc((size_t)TOK * DIN * 4);
    float*    mout_t = (float*)alloc((size_t)TOK * CDIM * 4);
    float*    of     = (float*)alloc((size_t)TOK * CDIM * 4);
    float*    ob     = (float*)alloc((size_t)TOK * CDIM * 4);
    float*    xres   = (float*)alloc((size_t)BB * TT * CDIM * HWC * 4);
    float*    gbuf   = (float*)alloc((size_t)BT * 32 * HWC * 4);
    float*    xw     = (float*)alloc((size_t)BB * TT * CDIM * HWC * 4);
    _Float16* cat16  = (_Float16*)alloc((size_t)TOK * DIN * 2);
    float*    projo  = (float*)alloc((size_t)TOK * CDIM * 4);

    // --- weight conversion (f32 -> f16, xproj zero-padded to XPD rows) ---
    for (int m = 0; m < 3; ++m) {
        cvt_f16_kernel<<<(512 * CDIM + 255) / 256, 256, 0, stream>>>(mw[m].in_w, in_w16[m], 512 * CDIM);
        pad_xproj_kernel<<<(XPD * DIN) / 256, 256, 0, stream>>>(mw[m].xproj_w, xproj16[m]);
        cvt_f16_kernel<<<(CDIM * DIN + 255) / 256, 256, 0, stream>>>(mw[m].out_w, out_w16[m], CDIM * DIN);
    }
    cvt_f16_kernel<<<(CDIM * DIN + 255) / 256, 256, 0, stream>>>(proj_w, proj_w16, CDIM * DIN);

    // --- temporal branch ---
    ln_temporal_kernel<<<TOK / 8, 256, 0, stream>>>(x, ln_t_g, ln_t_b, tokA);
    run_mamba(tokA, in_w16[0], xproj16[0], out_w16[0], mw[0],
              xz, xc, xc16, dbl, dtb, ybuf, mout_t, BHW, TT, stream);
    residual_t_kernel<<<(BB * TT * CDIM * HWC) / 256, 256, 0, stream>>>(x, mout_t, xres);

    // --- importance gate ---
    conv1_kernel<<<(BT * 32 * HWC) / 256, 256, 0, stream>>>(xres, g1_w, g1_b, gbuf);
    gate2_apply_kernel<<<(BT * HWC) / 256, 256, 0, stream>>>(gbuf, g2_w, g2_b, xres, xw);

    // --- spatial branch (fwd + bwd) ---
    ln_spatial_kernel<<<TOK / 8, 256, 0, stream>>>(xw, ln_s_g, ln_s_b, tokA, tokB);
    run_mamba(tokA, in_w16[1], xproj16[1], out_w16[1], mw[1],
              xz, xc, xc16, dbl, dtb, ybuf, of, BT, HWC, stream);
    run_mamba(tokB, in_w16[2], xproj16[2], out_w16[2], mw[2],
              xz, xc, xc16, dbl, dtb, ybuf, ob, BT, HWC, stream);

    // --- concat + final projection + residual ---
    concat_kernel<<<(TOK * DIN) / 256, 256, 0, stream>>>(of, ob, cat16);
    gemm_wmma_f16_kernel<<<dim3(GX, CDIM / 32), 128, 0, stream>>>(
        cat16, proj_w16, projo, TOK, CDIM, DIN);
    final_kernel<<<(BB * TT * CDIM * HWC) / 256, 256, 0, stream>>>(xres, projo, proj_b, out);
}
